// Attention_2087354105906
// MI455X (gfx1250) — matrix operand order repaired and checked by reading, not compiled, not run on hardware
//
#include <hip/hip_runtime.h>

typedef __attribute__((ext_vector_type(16))) _Float16 v16h;
typedef __attribute__((ext_vector_type(8)))  _Float16 v8h;
typedef __attribute__((ext_vector_type(8)))  float    v8f;

#define WMMA_F16(a, b, c) \
  __builtin_amdgcn_wmma_f32_16x16x32_f16(false, (a), false, (b), (short)0, (c), false, false)

// ---- constants for this problem size ----
#define CDIM 512
#define BATCH 4
#define HS 48
#define NPIX 2304      // 48*48
#define MS 625         // 25*25
#define MP 640         // padded M (40 tiles of 16)
#define NHEAD 8
#define HD 64

// A-matrix (16x32 f16) per-lane fragment: row = lane%16 of Arow base,
// element s -> K = k0 + (lane/16)*16 + s.
__device__ __forceinline__ v16h ld_a_frag(const _Float16* rowp, int k0, int lhalf) {
  v8h lo = *(const v8h*)(rowp + k0 + lhalf * 16);
  v8h hi = *(const v8h*)(rowp + k0 + lhalf * 16 + 8);
  v16h r;
#pragma unroll
  for (int i = 0; i < 8; ++i) { r[i] = lo[i]; r[i + 8] = hi[i]; }
  return r;
}

// ======================= prep kernels =======================
__global__ void cvt_f16_kernel(const float* __restrict__ s, _Float16* __restrict__ d, int n) {
  int i = blockIdx.x * 256 + threadIdx.x;
  if (i < n) d[i] = (_Float16)s[i];
}

__global__ void prep_sr2_kernel(const float* __restrict__ w, const float* __restrict__ g,
                                const float* __restrict__ bb, const float* __restrict__ m,
                                const float* __restrict__ v, _Float16* __restrict__ wo,
                                float* __restrict__ bias) {
  int i = blockIdx.x * 256 + threadIdx.x;
  if (i < CDIM * CDIM) {
    int o = i >> 9;
    float s = g[o] * rsqrtf(v[o] + 1e-5f);
    wo[i] = (_Float16)(w[i] * s);
    if (i < CDIM) bias[i] = bb[i] - m[i] * g[i] * rsqrtf(v[i] + 1e-5f);
  }
}

__global__ void prep_sr1_kernel(const float* __restrict__ w, const float* __restrict__ g,
                                const float* __restrict__ bb, const float* __restrict__ m,
                                const float* __restrict__ v, float* __restrict__ wo,
                                float* __restrict__ bias) {
  int i = blockIdx.x * 256 + threadIdx.x;
  if (i < CDIM * 4) {
    int c = i >> 2;
    float s = g[c] * rsqrtf(v[c] + 1e-5f);
    wo[i] = w[i] * s;
    if (i < CDIM) bias[i] = bb[i] - m[i] * g[i] * rsqrtf(v[i] + 1e-5f);
  }
}

// x[b][c][192][192] --(stride-4 nearest)--> xsT[b][n][c] f16  (n = h*48+w)
__global__ void downsample_kernel(const float* __restrict__ x, _Float16* __restrict__ xsT) {
  size_t idx = (size_t)blockIdx.x * 256 + threadIdx.x;   // B*NPIX*CDIM threads
  int c = idx & (CDIM - 1);
  size_t t = idx >> 9;
  int n = (int)(t % NPIX);
  int b = (int)(t / NPIX);
  int h = (n / HS) * 4, w = (n % HS) * 4;
  xsT[idx] = (_Float16)x[((size_t)b * CDIM + c) * (192 * 192) + (size_t)h * 192 + w];
}

// ======================= WMMA GEMM kernels =======================
// Q[n][o] = Wq[o][:] . xsT[n][:] + qb[o]   (per batch)
__global__ __launch_bounds__(128) void gemm_qproj_kernel(const _Float16* __restrict__ W,
                                                         const _Float16* __restrict__ X,
                                                         const float* __restrict__ qb,
                                                         _Float16* __restrict__ Q) {
  int wave = blockIdx.x * 4 + (threadIdx.x >> 5);
  int lane = threadIdx.x & 31, l16 = lane & 15, lhalf = lane >> 4;
  int b = wave / (32 * 36);
  int rem = wave % (32 * 36);
  int ot = rem / 36, ng = rem % 36;
  const _Float16* Arow = W + (size_t)(ot * 16 + l16) * CDIM;
  const _Float16* Bb = X + ((size_t)b * NPIX + ng * 64) * CDIM;
  v8f z = {0.f, 0.f, 0.f, 0.f, 0.f, 0.f, 0.f, 0.f};
  v8f acc[4] = {z, z, z, z};
  for (int k0 = 0; k0 < CDIM; k0 += 32) {
    v16h a = ld_a_frag(Arow, k0, lhalf);
#pragma unroll
    for (int t = 0; t < 4; ++t) {
      v16h bt = *(const v16h*)(Bb + (size_t)(t * 16 + l16) * CDIM + k0 + lhalf * 16);
      acc[t] = WMMA_F16(a, bt, acc[t]);
    }
  }
  int ob = ot * 16 + lhalf * 8;
#pragma unroll
  for (int t = 0; t < 4; ++t) {
    size_t n = (size_t)b * NPIX + ng * 64 + t * 16 + l16;
    v8h ov;
#pragma unroll
    for (int r = 0; r < 8; ++r) ov[r] = (_Float16)(acc[t][r] + qb[ob + r]);
    *(v8h*)(Q + n * CDIM + ob) = ov;
  }
}

// uT[m][o] = Wsr2s[o][:] . tT[m][:] + bias2[o]  (f32 out)
__global__ __launch_bounds__(128) void gemm_sr2_kernel(const _Float16* __restrict__ W,
                                                       const _Float16* __restrict__ X,
                                                       const float* __restrict__ bias,
                                                       float* __restrict__ U) {
  int wave = blockIdx.x * 4 + (threadIdx.x >> 5);
  int lane = threadIdx.x & 31, l16 = lane & 15, lhalf = lane >> 4;
  int b = wave / (32 * 10);
  int rem = wave % (32 * 10);
  int ot = rem / 10, ng = rem % 10;
  const _Float16* Arow = W + (size_t)(ot * 16 + l16) * CDIM;
  const _Float16* Bb = X + ((size_t)b * MP + ng * 64) * CDIM;
  v8f z = {0.f, 0.f, 0.f, 0.f, 0.f, 0.f, 0.f, 0.f};
  v8f acc[4] = {z, z, z, z};
  for (int k0 = 0; k0 < CDIM; k0 += 32) {
    v16h a = ld_a_frag(Arow, k0, lhalf);
#pragma unroll
    for (int t = 0; t < 4; ++t) {
      v16h bt = *(const v16h*)(Bb + (size_t)(t * 16 + l16) * CDIM + k0 + lhalf * 16);
      acc[t] = WMMA_F16(a, bt, acc[t]);
    }
  }
  int ob = ot * 16 + lhalf * 8;
#pragma unroll
  for (int t = 0; t < 4; ++t) {
    size_t m = (size_t)b * MP + ng * 64 + t * 16 + l16;
    v8f ov;
#pragma unroll
    for (int r = 0; r < 8; ++r) ov[r] = acc[t][r] + bias[ob + r];
    *(v8f*)(U + m * CDIM + ob) = ov;
  }
}

// kv[o][m] = Wkv[o][:] . kvT[m][:] + kvb[o]; o<512 -> kT[m][o] f16, o>=512 -> V[o-512][m] f16
__global__ __launch_bounds__(128) void gemm_kvproj_kernel(const _Float16* __restrict__ W,
                                                          const _Float16* __restrict__ X,
                                                          const float* __restrict__ kvb,
                                                          _Float16* __restrict__ Kt,
                                                          _Float16* __restrict__ V) {
  int wave = blockIdx.x * 4 + (threadIdx.x >> 5);
  int lane = threadIdx.x & 31, l16 = lane & 15, lhalf = lane >> 4;
  int b = wave / (64 * 10);
  int rem = wave % (64 * 10);
  int ot = rem / 10, ng = rem % 10;
  const _Float16* Arow = W + (size_t)(ot * 16 + l16) * CDIM;
  const _Float16* Bb = X + ((size_t)b * MP + ng * 64) * CDIM;
  v8f z = {0.f, 0.f, 0.f, 0.f, 0.f, 0.f, 0.f, 0.f};
  v8f acc[4] = {z, z, z, z};
  for (int k0 = 0; k0 < CDIM; k0 += 32) {
    v16h a = ld_a_frag(Arow, k0, lhalf);
#pragma unroll
    for (int t = 0; t < 4; ++t) {
      v16h bt = *(const v16h*)(Bb + (size_t)(t * 16 + l16) * CDIM + k0 + lhalf * 16);
      acc[t] = WMMA_F16(a, bt, acc[t]);
    }
  }
  int ob = ot * 16 + lhalf * 8;
  if (ot < 32) {  // K part -> transposed [m][c]
#pragma unroll
    for (int t = 0; t < 4; ++t) {
      size_t m = (size_t)b * MP + ng * 64 + t * 16 + l16;
      v8h ov;
#pragma unroll
      for (int r = 0; r < 8; ++r) ov[r] = (_Float16)(acc[t][r] + kvb[ob + r]);
      *(v8h*)(Kt + m * CDIM + ob) = ov;
    }
  } else {  // V part -> [c][m]
#pragma unroll
    for (int t = 0; t < 4; ++t) {
      int m = ng * 64 + t * 16 + l16;
#pragma unroll
      for (int r = 0; r < 8; ++r) {
        int o = ob + r;
        V[((size_t)b * CDIM + (o - CDIM)) * MP + m] = (_Float16)(acc[t][r] + kvb[o]);
      }
    }
  }
}

// ======================= depthwise convs =======================
// dw 2x2 s2 p1 (+folded BN1) + ReLU : xsT f16 -> tT[b][m][c] f16, zero padding m>=625
__global__ void dw1_kernel(const _Float16* __restrict__ xsT, const float* __restrict__ w,
                           const float* __restrict__ bias, _Float16* __restrict__ tT) {
  size_t idx = (size_t)blockIdx.x * 256 + threadIdx.x;  // B*MP*CDIM
  int c = idx & (CDIM - 1);
  size_t t = idx >> 9;
  int m = (int)(t % MP);
  int b = (int)(t / MP);
  float out = 0.f;
  if (m < MS) {
    int oh = m / 25, ow = m % 25;
    float acc = bias[c];
#pragma unroll
    for (int kh = 0; kh < 2; ++kh) {
      int ih = oh * 2 - 1 + kh;
      if (ih < 0 || ih >= HS) continue;
#pragma unroll
      for (int kw = 0; kw < 2; ++kw) {
        int iw = ow * 2 - 1 + kw;
        if (iw < 0 || iw >= HS) continue;
        acc += (float)xsT[((size_t)b * NPIX + ih * HS + iw) * CDIM + c] * w[c * 4 + kh * 2 + kw];
      }
    }
    out = fmaxf(acc, 0.f);
  }
  tT[idx] = (_Float16)out;
}

// dw 3x3 s1 p1 + bias + residual : uT f32 -> kvT[b][m][c] f16, zero padding
__global__ void lc_kernel(const float* __restrict__ U, const float* __restrict__ w,
                          const float* __restrict__ bias, _Float16* __restrict__ kvT) {
  size_t idx = (size_t)blockIdx.x * 256 + threadIdx.x;  // B*MP*CDIM
  int c = idx & (CDIM - 1);
  size_t t = idx >> 9;
  int m = (int)(t % MP);
  int b = (int)(t / MP);
  float out = 0.f;
  if (m < MS) {
    int y = m / 25, x = m % 25;
    float acc = bias[c] + U[idx];  // residual
#pragma unroll
    for (int dy = -1; dy <= 1; ++dy) {
      int yy = y + dy;
      if (yy < 0 || yy >= 25) continue;
#pragma unroll
      for (int dx = -1; dx <= 1; ++dx) {
        int xx = x + dx;
        if (xx < 0 || xx >= 25) continue;
        acc += U[((size_t)b * MP + yy * 25 + xx) * CDIM + c] * w[c * 9 + (dy + 1) * 3 + (dx + 1)];
      }
    }
    out = acc;
  }
  kvT[idx] = (_Float16)out;
}

// ======================= flash attention =======================
// one wave: (b, head, 16 query rows) -> 16x64 output tile; loop keys in chunks of 32
__global__ __launch_bounds__(128) void attn_kernel(const _Float16* __restrict__ Q,
                                                   const _Float16* __restrict__ Kt,
                                                   const _Float16* __restrict__ V,
                                                   float* __restrict__ O) {
  __shared__ float smem[4][16 * 32];
  int wid = threadIdx.x >> 5;
  float* sm = smem[wid];
  int wave = blockIdx.x * 4 + wid;
  int lane = threadIdx.x & 31, l16 = lane & 15, lhalf = lane >> 4;
  int b = wave / (NHEAD * 144);
  int rem = wave % (NHEAD * 144);
  int h = rem / 144, nT = rem % 144;

  const _Float16* qb_ = Q + ((size_t)b * NPIX + nT * 16) * CDIM + h * HD;
  const _Float16* kb_ = Kt + (size_t)b * MP * CDIM + h * HD;
  const _Float16* vb_ = V + ((size_t)b * CDIM + h * HD) * MP;

  const _Float16* qrow = qb_ + (size_t)l16 * CDIM;
  v16h aq0 = ld_a_frag(qrow, 0, lhalf);
  v16h aq1 = ld_a_frag(qrow, 32, lhalf);

  v8f z = {0.f, 0.f, 0.f, 0.f, 0.f, 0.f, 0.f, 0.f};
  v8f acc[4] = {z, z, z, z};
  float mrun[8], lrun[8], alpha[8];
#pragma unroll
  for (int r = 0; r < 8; ++r) { mrun[r] = -1e30f; lrun[r] = 0.f; }

  for (int m0 = 0; m0 < MP; m0 += 32) {
    v8f s[2] = {z, z};
#pragma unroll
    for (int sub = 0; sub < 2; ++sub) {
      const _Float16* krow = kb_ + (size_t)(m0 + sub * 16 + l16) * CDIM;
      v16h bk0 = *(const v16h*)(krow + 0 + lhalf * 16);
      v16h bk1 = *(const v16h*)(krow + 32 + lhalf * 16);
      s[sub] = WMMA_F16(aq0, bk0, s[sub]);
      s[sub] = WMMA_F16(aq1, bk1, s[sub]);
      int col = m0 + sub * 16 + l16;
#pragma unroll
      for (int r = 0; r < 8; ++r)
        s[sub][r] = (col < MS) ? s[sub][r] * 0.125f : -1e30f;
    }
    // online softmax: row max / rescale / exp / row sum (rows striped: r + lhalf*8)
#pragma unroll
    for (int r = 0; r < 8; ++r) {
      float v = fmaxf(s[0][r], s[1][r]);
      v = fmaxf(v, __shfl_xor(v, 1, 32));
      v = fmaxf(v, __shfl_xor(v, 2, 32));
      v = fmaxf(v, __shfl_xor(v, 4, 32));
      v = fmaxf(v, __shfl_xor(v, 8, 32));
      float mnew = fmaxf(mrun[r], v);
      alpha[r] = __expf(mrun[r] - mnew);
      mrun[r] = mnew;
    }
#pragma unroll
    for (int sub = 0; sub < 2; ++sub)
#pragma unroll
      for (int r = 0; r < 8; ++r) s[sub][r] = __expf(s[sub][r] - mrun[r]);
#pragma unroll
    for (int r = 0; r < 8; ++r) {
      float rs = s[0][r] + s[1][r];
      rs += __shfl_xor(rs, 1, 32);
      rs += __shfl_xor(rs, 2, 32);
      rs += __shfl_xor(rs, 4, 32);
      rs += __shfl_xor(rs, 8, 32);
      lrun[r] = lrun[r] * alpha[r] + rs;
#pragma unroll
      for (int t = 0; t < 4; ++t) acc[t][r] *= alpha[r];
    }
    // stage P through LDS: D-layout (rows striped over lanes) -> A-layout (16x32 f16)
#pragma unroll
    for (int sub = 0; sub < 2; ++sub)
#pragma unroll
      for (int r = 0; r < 8; ++r)
        sm[(r + lhalf * 8) * 32 + sub * 16 + l16] = s[sub][r];
    asm volatile("s_wait_dscnt 0" ::: "memory");
    v16h pA;
    const float* prow = sm + l16 * 32 + lhalf * 16;
#pragma unroll
    for (int j = 0; j < 8; ++j) {
      pA[j] = (_Float16)prow[j];
      pA[8 + j] = (_Float16)prow[8 + j];
    }
    asm volatile("s_wait_dscnt 0" ::: "memory");
#pragma unroll
    for (int t = 0; t < 4; ++t) {
      v16h bv = *(const v16h*)(vb_ + (size_t)(t * 16 + l16) * MP + m0 + lhalf * 16);
      acc[t] = WMMA_F16(pA, bv, acc[t]);
    }
  }
  // normalize + write oT[b][n][c]
#pragma unroll
  for (int t = 0; t < 4; ++t) {
#pragma unroll
    for (int r = 0; r < 8; ++r) {
      int n = nT * 16 + r + lhalf * 8;
      int c = h * HD + t * 16 + l16;
      O[((size_t)b * NPIX + n) * CDIM + c] = acc[t][r] / lrun[r];
    }
  }
}

// oT[b][n][c] -> out[b][c][192][192], nearest x4 (float4 per thread, coalesced)
__global__ void upsample_kernel(const float* __restrict__ O, float* __restrict__ out) {
  size_t idx = (size_t)blockIdx.x * 256 + threadIdx.x;  // B*CDIM*192*48
  int w4 = (int)(idx % 48);
  size_t t = idx / 48;
  int hh = (int)(t % 192);
  t /= 192;
  int c = (int)(t % CDIM);
  int b = (int)(t / CDIM);
  float v = O[((size_t)b * NPIX + (hh >> 2) * HS + w4) * CDIM + c];
  float4 f = make_float4(v, v, v, v);
  ((float4*)out)[idx] = f;
}

// ======================= host launch =======================
extern "C" void kernel_launch(void* const* d_in, const int* in_sizes, int n_in,
                              void* d_out, int out_size, void* d_ws, size_t ws_size,
                              hipStream_t stream) {
  const float* x     = (const float*)d_in[0];
  const float* q_w   = (const float*)d_in[1];
  const float* q_b   = (const float*)d_in[2];
  const float* kv_w  = (const float*)d_in[3];
  const float* kv_b  = (const float*)d_in[4];
  const float* sr1_w = (const float*)d_in[5];
  const float* bn1_g = (const float*)d_in[6];
  const float* bn1_b = (const float*)d_in[7];
  const float* bn1_m = (const float*)d_in[8];
  const float* bn1_v = (const float*)d_in[9];
  const float* sr2_w = (const float*)d_in[10];
  const float* bn2_g = (const float*)d_in[11];
  const float* bn2_b = (const float*)d_in[12];
  const float* bn2_m = (const float*)d_in[13];
  const float* bn2_v = (const float*)d_in[14];
  const float* lc_w  = (const float*)d_in[15];
  const float* lc_b  = (const float*)d_in[16];
  float* out = (float*)d_out;

  char* w = (char*)d_ws;
  size_t off = 0;
  auto alloc = [&](size_t bytes) {
    void* p = w + off;
    off = (off + bytes + 255) & ~(size_t)255;
    return p;
  };
  _Float16* wq16  = (_Float16*)alloc((size_t)CDIM * CDIM * 2);
  _Float16* wkv16 = (_Float16*)alloc((size_t)2 * CDIM * CDIM * 2);
  _Float16* wsr2  = (_Float16*)alloc((size_t)CDIM * CDIM * 2);
  float* sr1ws    = (float*)alloc((size_t)CDIM * 4 * 4);
  float* bias1    = (float*)alloc((size_t)CDIM * 4);
  float* bias2    = (float*)alloc((size_t)CDIM * 4);
  _Float16* xsT   = (_Float16*)alloc((size_t)BATCH * NPIX * CDIM * 2);
  _Float16* qT    = (_Float16*)alloc((size_t)BATCH * NPIX * CDIM * 2);
  _Float16* tT    = (_Float16*)alloc((size_t)BATCH * MP * CDIM * 2);
  float* uT       = (float*)alloc((size_t)BATCH * MP * CDIM * 4);
  _Float16* kvT   = (_Float16*)alloc((size_t)BATCH * MP * CDIM * 2);
  _Float16* kT    = (_Float16*)alloc((size_t)BATCH * MP * CDIM * 2);
  _Float16* vv    = (_Float16*)alloc((size_t)BATCH * CDIM * MP * 2);
  float* oT       = (float*)alloc((size_t)BATCH * NPIX * CDIM * 4);

  cvt_f16_kernel<<<(CDIM * CDIM + 255) / 256, 256, 0, stream>>>(q_w, wq16, CDIM * CDIM);
  cvt_f16_kernel<<<(2 * CDIM * CDIM + 255) / 256, 256, 0, stream>>>(kv_w, wkv16, 2 * CDIM * CDIM);
  prep_sr2_kernel<<<(CDIM * CDIM + 255) / 256, 256, 0, stream>>>(sr2_w, bn2_g, bn2_b, bn2_m, bn2_v, wsr2, bias2);
  prep_sr1_kernel<<<(CDIM * 4 + 255) / 256, 256, 0, stream>>>(sr1_w, bn1_g, bn1_b, bn1_m, bn1_v, sr1ws, bias1);

  downsample_kernel<<<(BATCH * NPIX * CDIM) / 256, 256, 0, stream>>>(x, xsT);

  gemm_qproj_kernel<<<(BATCH * 32 * 36) / 4, 128, 0, stream>>>(wq16, xsT, q_b, qT);

  dw1_kernel<<<(BATCH * MP * CDIM) / 256, 256, 0, stream>>>(xsT, sr1ws, bias1, tT);
  gemm_sr2_kernel<<<(BATCH * 32 * 10) / 4, 128, 0, stream>>>(wsr2, tT, bias2, uT);
  lc_kernel<<<(BATCH * MP * CDIM) / 256, 256, 0, stream>>>(uT, lc_w, lc_b, kvT);
  gemm_kvproj_kernel<<<(BATCH * 64 * 10) / 4, 128, 0, stream>>>(wkv16, kvT, kv_b, kT, vv);

  attn_kernel<<<(BATCH * NHEAD * 144) / 4, 128, 0, stream>>>(qT, kT, vv, oT);

  upsample_kernel<<<((size_t)BATCH * CDIM * 192 * 48) / 256, 256, 0, stream>>>(oT, out);
}
